// DensePoseV1ConvXGNSparseGNHead_79207786872964
// MI455X (gfx1250) — matrix operand order repaired and checked by reading, not compiled, not run on hardware
//
#include <hip/hip_runtime.h>
#include <hip/hip_bf16.h>

typedef __attribute__((ext_vector_type(16))) __bf16 v16bf;
typedef __attribute__((ext_vector_type(8)))  __bf16 v8bf;
typedef __attribute__((ext_vector_type(8)))  float  v8f;
typedef __attribute__((ext_vector_type(4)))  int    v4i;
typedef __attribute__((ext_vector_type(8)))  int    v8i;

#define CCH     256
#define GSZ     8          // channels per group (256/32)
#define NUM_INS 64
#define EPSV    1e-5f
#define KECA    3
#define USE_TDM   1        // gfx1250 tensor_load_to_lds for B tiles
#define USE_ASYNC 1        // fallback: global_load_async_to_lds_b128

// Tile config: 64 (M) x 128 (N) x 32 (K), 128 threads = 4 waves,
// each wave computes 16x128 via 8 v_wmma_f32_16x16x32_bf16 accumulators.
#define BM 64
#define BN 128
#define BK 32
#define APAD 40            // LDS row stride in bf16 (80 B, 16-B aligned)
#define CPAD 132           // f32 C-staging row stride

#define SM_TILES  ((BM + BN) * APAD * 2)     // As + BsT bytes (15360)
#define SM_CSTAGE (BM * CPAD * 4)            // 33792
#define SM_BYTES  (SM_CSTAGE > SM_TILES ? SM_CSTAGE : SM_TILES)

union FragAB { v16bf v; v8bf h[2]; };

// ---------------------------------------------------------------------------
// One-time weight prep: Wt[l][n][k] = bf16(W[l][k][n])   (B operand, column-
// major as BsT rows -> hot-loop B staging becomes a pure tile copy).
// ---------------------------------------------------------------------------
__global__ __launch_bounds__(256)
void wprep_kernel(const float* __restrict__ W, __bf16* __restrict__ Wt)
{
    const int ln = blockIdx.x;            // grid = L*CCH, one block per (l,n)
    const int l  = ln >> 8, n = ln & 255;
    const int k  = threadIdx.x;
    const float v = W[(size_t)l * CCH * CCH + (size_t)k * CCH + n];
    Wt[(size_t)l * CCH * CCH + (size_t)n * CCH + k] = (__bf16)v;
}

// ---------------------------------------------------------------------------
__global__ __launch_bounds__(128)
void gemm_act_kernel(const float*  __restrict__ x0,     // layer-0 f32 input
                     const __bf16* __restrict__ yprev,  // bf16 prev-layer raw y
                     const float*  __restrict__ scl,    // [I,C] fused scale
                     const float*  __restrict__ bia,    // [I,C] fused bias
                     const int*    __restrict__ ids,    // [N]
                     const __bf16* __restrict__ Wtl,    // [C,C] bf16, (n,k) layout
                     const float*  __restrict__ bl,     // [C]
                     __bf16*       __restrict__ yout,   // [N,C] bf16
                     int Ntok, int first)
{
    __shared__ __align__(16) char smem[SM_BYTES];
    __bf16 (*As )[APAD] = (__bf16(*)[APAD])smem;                      // [BM][APAD]
    __bf16 (*BsT)[APAD] = (__bf16(*)[APAD])(smem + BM * APAD * 2);    // [BN][APAD]
    float  (*Cs )[CPAD] = (float (*)[CPAD])smem;                      // reused post-loop

    const int tid  = threadIdx.x;
    const int wave = tid >> 5;
    const int lane = tid & 31;
    const int lr   = lane & 15;
    const int hi   = lane >> 4;
    const int mBase = blockIdx.x * BM;
    const int nBase = blockIdx.y * BN;

    v8f acc[8] = {};

    // A staging: 2 threads per row, 16 consecutive k each
    const int ar = tid >> 1;
    const int ak = (tid & 1) * 16;
    const int am = mBase + ar;
    int ains = 0;
    if (!first && am < Ntok) ains = ids[am];

#if USE_TDM
    // Tensor DMA descriptor (ISA ch.8): 2-D bf16 tensor Wt[C][C], tile 32(k) x
    // 128(n) into padded LDS rows (64 B data + 16 B pad == APAD).  Group 1 is
    // loop-invariant; group 0 carries lds/global addresses (uniform -> SGPRs).
    const unsigned bLdsBase = (unsigned)(uintptr_t)&BsT[0][0];
    const unsigned long long gBase =
        (unsigned long long)(uintptr_t)Wtl + (unsigned long long)nBase * (CCH * 2u);
    v8i tg1;
    tg1[0] = (1 << 16)        // data_size = 2 bytes
           | (1 << 20)        // pad_enable
           | (3 << 22)        // pad_interval: 16 DWORDs (64 B) between pads
           | (3 << 25);       // pad_amount:   4 DWORDs (16 B)
    tg1[1] = (CCH & 0xFFFF) << 16;          // tensor_dim0 = 256 (lo16)
    tg1[2] = (CCH & 0xFFFF) << 16;          // tensor_dim0 hi=0 | tensor_dim1 lo16
    tg1[3] = (BK << 16);                    // tensor_dim1 hi=0 | tile_dim0 = 32
    tg1[4] = BN;                            // tile_dim1 = 128, tile_dim2 = 0
    tg1[5] = CCH;                           // tensor_dim0_stride = 256 (lo32)
    tg1[6] = 0;                             // stride hi | tensor_dim1_stride lo
    tg1[7] = 0;
#elif USE_ASYNC
    const unsigned bLds  = (unsigned)(uintptr_t)&BsT[tid][0];         // LDS byte addr
    const unsigned bGofs = (unsigned)(nBase + tid) * (CCH * 2u);      // row byte offset
#endif

    for (int k0 = 0; k0 < CCH; k0 += BK) {
        __syncthreads();
        // ---- stage B tile ----
#if USE_TDM
        if (wave == 0) {
            const unsigned long long ga = gBase + (unsigned)k0 * 2u;
            v4i tg0;
            tg0[0] = 1;                              // count=1 valid descriptor
            tg0[1] = (int)bLdsBase;                  // lds_addr
            tg0[2] = (int)(unsigned)ga;              // global_addr lo
            tg0[3] = (int)((unsigned)(ga >> 32) | (2u << 30));   // addr hi | type=2
            asm volatile("tensor_load_to_lds %0, %1"
                         :: "s"(tg0), "s"(tg1) : "memory");
        }
#elif USE_ASYNC
        {
            const unsigned go = bGofs + (unsigned)k0 * 2u;
            asm volatile(
                "global_load_async_to_lds_b128 %0, %1, %2\n\t"
                "global_load_async_to_lds_b128 %0, %1, %2 offset:16\n\t"
                "global_load_async_to_lds_b128 %0, %1, %2 offset:32\n\t"
                "global_load_async_to_lds_b128 %0, %1, %2 offset:48"
                :: "v"(bLds), "v"(go), "s"(Wtl) : "memory");
        }
#else
        {
            const v8bf* src = (const v8bf*)((const char*)Wtl +
                              (size_t)(nBase + tid) * (CCH * 2u) + (size_t)k0 * 2u);
            *(v8bf*)&BsT[tid][0]  = src[0];
            *(v8bf*)&BsT[tid][8]  = src[1];
            *(v8bf*)&BsT[tid][16] = src[2];
            *(v8bf*)&BsT[tid][24] = src[3];
        }
#endif
        // ---- stage A tile (apply prev-layer affine + relu unless first) ----
        {
            union { v8bf h[2]; __bf16 e[16]; } aw;
            if (am < Ntok) {
                if (first) {
                    const float4* s = (const float4*)(x0 + (size_t)am * CCH + k0 + ak);
                    #pragma unroll
                    for (int v = 0; v < 4; ++v) {
                        float4 f = s[v];
                        aw.e[4*v + 0] = (__bf16)f.x;
                        aw.e[4*v + 1] = (__bf16)f.y;
                        aw.e[4*v + 2] = (__bf16)f.z;
                        aw.e[4*v + 3] = (__bf16)f.w;
                    }
                } else {
                    union { v8bf h[2]; __bf16 e[16]; } yv;
                    const __bf16* s = yprev + (size_t)am * CCH + k0 + ak;
                    yv.h[0] = *(const v8bf*)(s);
                    yv.h[1] = *(const v8bf*)(s + 8);
                    const float4* sc4 = (const float4*)(scl + ains * CCH + k0 + ak);
                    const float4* bi4 = (const float4*)(bia + ains * CCH + k0 + ak);
                    #pragma unroll
                    for (int v = 0; v < 4; ++v) {
                        float4 sc = sc4[v], bi = bi4[v];
                        float f0 = (float)yv.e[4*v + 0] * sc.x + bi.x;
                        float f1 = (float)yv.e[4*v + 1] * sc.y + bi.y;
                        float f2 = (float)yv.e[4*v + 2] * sc.z + bi.z;
                        float f3 = (float)yv.e[4*v + 3] * sc.w + bi.w;
                        aw.e[4*v + 0] = (__bf16)fmaxf(f0, 0.0f);
                        aw.e[4*v + 1] = (__bf16)fmaxf(f1, 0.0f);
                        aw.e[4*v + 2] = (__bf16)fmaxf(f2, 0.0f);
                        aw.e[4*v + 3] = (__bf16)fmaxf(f3, 0.0f);
                    }
                }
            } else {
                #pragma unroll
                for (int v = 0; v < 16; ++v) aw.e[v] = (__bf16)0.0f;
            }
            *(v8bf*)&As[ar][ak]     = aw.h[0];
            *(v8bf*)&As[ar][ak + 8] = aw.h[1];
        }
        // ---- prefetch next A k-tile (global_prefetch_b8) ----
        if (k0 + BK < CCH && am < Ntok) {
            if (first) __builtin_prefetch(x0    + (size_t)am * CCH + k0 + BK + ak, 0, 3);
            else       __builtin_prefetch(yprev + (size_t)am * CCH + k0 + BK + ak, 0, 3);
        }
#if USE_TDM
        if (wave == 0)
            asm volatile("s_wait_tensorcnt 0x0" ::: "memory");  // TDM done -> LDS
#elif USE_ASYNC
        asm volatile("s_wait_asynccnt 0x0" ::: "memory");       // drain async copy
#endif
        __syncthreads();

        // ---- fragments: ISA 7.12.2 layout == two contiguous 16-B runs ----
        FragAB a;
        {
            const __bf16* arow = &As[wave * 16 + lr][0];
            a.h[0] = *(const v8bf*)(arow + hi * 8);          // k = hi*8 .. +7
            a.h[1] = *(const v8bf*)(arow + 16 + hi * 8);     // k = 16+hi*8 .. +7
        }
        FragAB bf[8];
        #pragma unroll
        for (int t = 0; t < 8; ++t) {                        // preload all B frags
            const __bf16* brow = &BsT[t * 16 + lr][0];
            bf[t].h[0] = *(const v8bf*)(brow + hi * 8);      // k = hi*8 .. +7
            bf[t].h[1] = *(const v8bf*)(brow + 16 + hi * 8); // k = 16+hi*8 .. +7
        }
        #pragma unroll
        for (int t = 0; t < 8; ++t)                          // back-to-back WMMAs
            acc[t] = __builtin_amdgcn_wmma_f32_16x16x32_bf16(
                         false, a.v, false, bf[t].v, (short)0, acc[t], false, false);
    }

    // ---- epilogue: stage C through LDS for coalesced b128 stores ----
    __syncthreads();                      // done reading As/BsT (Cs aliases them)
    {
        const int crow = wave * 16;
        #pragma unroll
        for (int t = 0; t < 8; ++t)
            #pragma unroll
            for (int r = 0; r < 8; ++r)
                Cs[crow + r + hi * 8][t * 16 + lr] = acc[t][r];
    }
    __syncthreads();
    {
        const int row  = tid >> 1;
        const int c0   = (tid & 1) * 64;
        const int grow = mBase + row;
        if (grow < Ntok) {
            #pragma unroll
            for (int q = 0; q < 8; ++q) {
                union { v8bf h; __bf16 e[8]; } ow;
                #pragma unroll
                for (int e = 0; e < 8; ++e) {
                    const int n = c0 + q * 8 + e;
                    ow.e[e] = (__bf16)(Cs[row][n] + bl[nBase + n]);
                }
                *(v8bf*)(yout + (size_t)grow * CCH + nBase + c0 + q * 8) = ow.h;
            }
        }
    }
}

// ---------------------------------------------------------------------------
// Per-(instance,channel) segment sum / sum-of-squares.  One thread per
// channel, serial token loop -> conflict-free LDS accumulation (128 KB
// dynamic LDS; CDNA5 allows up to 320 KB per workgroup), then global atomics.
// ---------------------------------------------------------------------------
__global__ __launch_bounds__(256)
void stats_kernel(const __bf16* __restrict__ y, const int* __restrict__ ids,
                  float* __restrict__ S, float* __restrict__ S2,
                  int Ntok, int tokPerBlk)
{
    extern __shared__ float sm[];            // NUM_INS*CCH*2 floats
    float* lS  = sm;
    float* lS2 = sm + NUM_INS * CCH;
    const int c = threadIdx.x;

    for (int i = c; i < NUM_INS * CCH * 2; i += 256) sm[i] = 0.0f;
    __syncthreads();

    const int t0 = blockIdx.x * tokPerBlk;
    const int t1 = min(t0 + tokPerBlk, Ntok);
    for (int t = t0; t < t1; ++t) {
        const int ins = ids[t];
        const float v = (float)y[(size_t)t * CCH + c];
        lS [ins * CCH + c] += v;
        lS2[ins * CCH + c] += v * v;
    }
    __syncthreads();

    for (int i = c; i < NUM_INS * CCH; i += 256) {
        atomicAdd(&S[i],  lS[i]);
        atomicAdd(&S2[i], lS2[i]);
    }
}

// ---------------------------------------------------------------------------
// Fold GroupNorm + ECA gate into per-(instance,channel) scale/bias:
//   out = relu(y * A + B),  A = inv*gamma*gate,  B = (beta - mean*inv*gamma)*gate
// ---------------------------------------------------------------------------
__global__ __launch_bounds__(256)
void postnorm_kernel(const float* __restrict__ S,  const float* __restrict__ S2,
                     const float* __restrict__ counts,
                     const float* __restrict__ gamma, const float* __restrict__ beta,
                     const float* __restrict__ eca,
                     float* __restrict__ scl, float* __restrict__ bia)
{
    __shared__ float sS[CCH], sS2[CCH], smv[CCH + 2];
    const int i = blockIdx.x;
    const int c = threadIdx.x;

    const float cnt = fmaxf(counts[i], 1.0f);
    const float sv  = S [i * CCH + c];
    const float s2v = S2[i * CCH + c];
    sS[c] = sv; sS2[c] = s2v;
    if (c == 0) { smv[0] = 0.0f; smv[CCH + 1] = 0.0f; }
    __syncthreads();

    const int g0 = (c / GSZ) * GSZ;
    float gs = 0.0f, gs2 = 0.0f;
    #pragma unroll
    for (int j = 0; j < GSZ; ++j) { gs += sS[g0 + j]; gs2 += sS2[g0 + j]; }

    const float denom = cnt * (float)GSZ;
    const float mean  = gs / denom;
    const float var   = gs2 / denom - mean * mean;
    const float inv   = rsqrtf(var + EPSV);
    const float cm    = sv / cnt;                 // per-channel segment mean
    const float ga    = gamma[c], be = beta[c];

    smv[c + 1] = (cm - mean) * inv * ga + be;     // ECA input m_c
    __syncthreads();

    const float conv = eca[0] * smv[c] + eca[1] * smv[c + 1] + eca[2] * smv[c + 2];
    const float gate = 1.0f / (1.0f + __expf(-conv));

    scl[i * CCH + c] = inv * ga * gate;
    bia[i * CCH + c] = (be - mean * inv * ga) * gate;
}

__global__ void counts_kernel(const int* __restrict__ ids, float* __restrict__ counts,
                              int Ntok)
{
    const int t = blockIdx.x * 256 + threadIdx.x;
    if (t < Ntok) atomicAdd(&counts[ids[t]], 1.0f);
}

// Apply the final layer's affine+relu and emit f32 output.
__global__ __launch_bounds__(256)
void finalize_kernel(const __bf16* __restrict__ y, const int* __restrict__ ids,
                     const float* __restrict__ scl, const float* __restrict__ bia,
                     float* __restrict__ out, int Ntok)
{
    const int t = blockIdx.x;
    const int c = threadIdx.x;
    if (t >= Ntok) return;
    const int ins = ids[t];
    const float v = (float)y[(size_t)t * CCH + c] * scl[ins * CCH + c] + bia[ins * CCH + c];
    out[(size_t)t * CCH + c] = fmaxf(v, 0.0f);
}

// ---------------------------------------------------------------------------
extern "C" void kernel_launch(void* const* d_in, const int* in_sizes, int n_in,
                              void* d_out, int out_size, void* d_ws, size_t ws_size,
                              hipStream_t stream)
{
    const float* features = (const float*)d_in[0];
    const float* W        = (const float*)d_in[1];  // [L,C,C]
    const float* b        = (const float*)d_in[2];  // [L,C]
    const float* gamma    = (const float*)d_in[3];  // [L,C]
    const float* beta     = (const float*)d_in[4];  // [L,C]
    const float* eca      = (const float*)d_in[5];  // [L,K]
    const int*   ids      = (const int*)d_in[6];    // [N]
    const int Ntok = in_sizes[0] / CCH;
    const int L = 8;

    char* ws = (char*)d_ws;
    size_t ybytes = ((size_t)Ntok * CCH * sizeof(__bf16) + 255) & ~(size_t)255;
    __bf16* bufs[2] = { (__bf16*)ws, (__bf16*)(ws + ybytes) };
    float* counts = (float*)(ws + 2 * ybytes);
    float* S      = counts + 256;           // counts padded to 256 floats
    float* S2     = S  + NUM_INS * CCH;     // S,S2 contiguous for one memset
    float* scl    = S2 + NUM_INS * CCH;
    float* bia    = scl + NUM_INS * CCH;
    __bf16* Wt    = (__bf16*)(bia + NUM_INS * CCH);   // [L,C,C] bf16 transposed

    wprep_kernel<<<L * CCH, 256, 0, stream>>>(W, Wt);

    (void)hipMemsetAsync(counts, 0, 256 * sizeof(float), stream);
    counts_kernel<<<(Ntok + 255) / 256, 256, 0, stream>>>(ids, counts, Ntok);

    const dim3 ggrid((Ntok + BM - 1) / BM, CCH / BN);
    const int tokPerBlk = 1024;
    const int nStat = (Ntok + tokPerBlk - 1) / tokPerBlk;
    const size_t statLds = (size_t)NUM_INS * CCH * 2 * sizeof(float);   // 128 KB

    for (int l = 0; l < L; ++l) {
        __bf16* yo = bufs[l & 1];
        const __bf16* yp = (l == 0) ? (const __bf16*)nullptr : bufs[(l - 1) & 1];
        gemm_act_kernel<<<ggrid, 128, 0, stream>>>(
            features, yp, scl, bia, ids,
            Wt + (size_t)l * CCH * CCH, b + (size_t)l * CCH, yo, Ntok, l == 0 ? 1 : 0);

        (void)hipMemsetAsync(S, 0, 2 * NUM_INS * CCH * sizeof(float), stream);
        stats_kernel<<<nStat, 256, statLds, stream>>>(yo, ids, S, S2, Ntok, tokPerBlk);

        postnorm_kernel<<<NUM_INS, 256, 0, stream>>>(
            S, S2, counts, gamma + l * CCH, beta + l * CCH, eca + l * KECA, scl, bia);
    }

    finalize_kernel<<<Ntok, 256, 0, stream>>>(bufs[(L - 1) & 1], ids, scl, bia,
                                              (float*)d_out, Ntok);
    (void)n_in; (void)out_size; (void)ws_size;
}
